// GRUEncoderDecoder_22548578304258
// MI455X (gfx1250) — compile-verified, matrix operand order checked by reading, not run on hardware
//
#include <hip/hip_runtime.h>
#include <math.h>

// ---------------------------------------------------------------------------
// Pointer-generator GRU encoder-decoder for MI455X (gfx1250, wave32, WMMA).
// All GEMMs use v_wmma_f32_16x16x32_bf16 with fp32 accumulate; weights are
// converted once (fp32 -> bf16, transposed to [out,in]) into workspace so that
// per-lane WMMA A/B fragments are contiguous 32-byte loads.
// GRU K-loops use "#pragma unroll 1" so fragments are consumed right after load
// (previous revision fully unrolled and spilled fragments to scratch).
// ---------------------------------------------------------------------------

typedef __bf16 bf16;
typedef __attribute__((ext_vector_type(16))) __bf16 bf16x16;
typedef __attribute__((ext_vector_type(8)))  float  f32x8;

#define VSZ 50000
#define OOVSZ 50
#define ESZ 256
#define HEN 512
#define HDN 512
#define NB 32
#define SL 400
#define TDL 30
#define VOUT (VSZ + OOVSZ)   // 50050

#define DEV static __device__ __forceinline__

DEV f32x8 wmma_bf(bf16x16 a, bf16x16 b, f32x8 c) {
  // D = A(16x32 bf16) x B(32x16 bf16) + C(16x16 f32)
  return __builtin_amdgcn_wmma_f32_16x16x32_bf16(false, a, false, b,
                                                 (short)0, c, false, false);
}
DEV f32x8 zero8() { f32x8 z = {0.f,0.f,0.f,0.f,0.f,0.f,0.f,0.f}; return z; }
DEV bf16x16 ldbf(const bf16* p) { return *(const bf16x16*)p; }
DEV bf16x16 cvtf16(const float* p) {
  bf16x16 r;
#pragma unroll
  for (int i = 0; i < 16; ++i) r[i] = (bf16)p[i];
  return r;
}
DEV float sigf(float x) { return 1.0f / (1.0f + __expf(-x)); }

// ---------------------------------------------------------------------------
// Weight conversion kernels (one-time per launch)
// ---------------------------------------------------------------------------
__global__ void cvt_bf16_kernel(const float* __restrict__ in,
                                bf16* __restrict__ out, int count) {
  int i = blockIdx.x * 256 + threadIdx.x;
  if (i < count) out[i] = (bf16)in[i];
}

// in: [cols_out, rows_out] row-major  ->  out: [rows_out, cols_out] row-major
__global__ void cvt_t_bf16_kernel(const float* __restrict__ in,
                                  bf16* __restrict__ out,
                                  int rows_out, int cols_out) {
  int i = blockIdx.x * 256 + threadIdx.x;
  if (i < rows_out * cols_out) {
    int r = i / cols_out, c = i % cols_out;
    out[i] = (bf16)in[(size_t)c * rows_out + r];
  }
}

__global__ void zero_u32_kernel(unsigned int* p, int count) {
  int i = blockIdx.x * 256 + threadIdx.x;
  if (i < count) p[i] = 0u;
}

// ---------------------------------------------------------------------------
// Bidirectional GRU encoder: persistent kernel, grid = 2 blocks (fwd/bwd),
// 512 threads = 16 waves.  h double-buffered in LDS (fp32 + bf16 mirror).
// Each wave owns 2 gate-column tiles; both batch M-tiles are computed together
// so every weight fragment feeds two WMMAs.
// ---------------------------------------------------------------------------
__global__ __launch_bounds__(512)
void encoder_kernel(const int* __restrict__ src_ids,
                    const bf16* __restrict__ embed_bf,
                    const bf16* __restrict__ WihF, const bf16* __restrict__ WhhF,
                    const float* __restrict__ bihF, const float* __restrict__ bhhF,
                    const bf16* __restrict__ WihB, const bf16* __restrict__ WhhB,
                    const float* __restrict__ bihB, const float* __restrict__ bhhB,
                    float* __restrict__ enc) {
  const int dir = blockIdx.x;
  const bf16* Wih = dir ? WihB : WihF;
  const bf16* Whh = dir ? WhhB : WhhF;
  const float* bih = dir ? bihB : bihF;
  const float* bhh = dir ? bhhB : bhhF;

  __shared__ __align__(32) float h32[2][NB][HEN];   // 128 KB
  __shared__ __align__(32) bf16  hbf[2][NB][HEN];   //  64 KB
  __shared__ __align__(32) bf16  xbf[NB][ESZ];      //  16 KB

  const int tid = threadIdx.x;
  const int lane = tid & 31, wave = tid >> 5;
  const int lr = lane & 15;
  const int koff = (lane >> 4) << 4;
  const int mbase = (lane >> 4) << 3;

  for (int i = tid; i < NB * HEN; i += 512) {
    h32[0][i >> 9][i & 511] = 0.f;
    hbf[0][i >> 9][i & 511] = (bf16)0.f;
  }

  for (int t = 0; t < SL; ++t) {
    const int cur = t & 1, nxt = cur ^ 1;
    const int p = dir ? (SL - 1 - t) : t;
    {  // gather x_t = embed_bf[src_ids[:, p]]  (32x256 bf16)
      int n = tid >> 4;
      int c0 = (tid & 15) << 4;
      int id = src_ids[n * SL + p];
      *(bf16x16*)&xbf[n][c0] = ldbf(embed_bf + (size_t)id * ESZ + c0);
    }
    __syncthreads();

#pragma unroll 1
    for (int jj = 0; jj < 2; ++jj) {
      const int jt = wave * 2 + jj;
      const int j0 = jt * 16;
      f32x8 aR0 = zero8(), aZ0 = zero8(), aI0 = zero8(), aH0 = zero8();
      f32x8 aR1 = zero8(), aZ1 = zero8(), aI1 = zero8(), aH1 = zero8();

      const bf16* a0 = &xbf[lr][koff];
      const bf16* a1 = &xbf[16 + lr][koff];
      const bf16* wr = Wih + (size_t)(j0 + lr) * ESZ + koff;
#pragma unroll 1
      for (int k0 = 0; k0 < ESZ; k0 += 32) {
        bf16x16 br_ = ldbf(wr + k0);
        bf16x16 bz_ = ldbf(wr + (size_t)HEN * ESZ + k0);
        bf16x16 bi_ = ldbf(wr + (size_t)2 * HEN * ESZ + k0);
        bf16x16 A0 = ldbf(a0 + k0);
        bf16x16 A1 = ldbf(a1 + k0);
        aR0 = wmma_bf(A0, br_, aR0);  aR1 = wmma_bf(A1, br_, aR1);
        aZ0 = wmma_bf(A0, bz_, aZ0);  aZ1 = wmma_bf(A1, bz_, aZ1);
        aI0 = wmma_bf(A0, bi_, aI0);  aI1 = wmma_bf(A1, bi_, aI1);
      }
      const bf16* h0p = &hbf[cur][lr][koff];
      const bf16* h1p = &hbf[cur][16 + lr][koff];
      const bf16* wh = Whh + (size_t)(j0 + lr) * HEN + koff;
#pragma unroll 1
      for (int k0 = 0; k0 < HEN; k0 += 32) {
        bf16x16 br_ = ldbf(wh + k0);
        bf16x16 bz_ = ldbf(wh + (size_t)HEN * HEN + k0);
        bf16x16 bh_ = ldbf(wh + (size_t)2 * HEN * HEN + k0);
        bf16x16 A0 = ldbf(h0p + k0);
        bf16x16 A1 = ldbf(h1p + k0);
        aR0 = wmma_bf(A0, br_, aR0);  aR1 = wmma_bf(A1, br_, aR1);
        aZ0 = wmma_bf(A0, bz_, aZ0);  aZ1 = wmma_bf(A1, bz_, aZ1);
        aH0 = wmma_bf(A0, bh_, aH0);  aH1 = wmma_bf(A1, bh_, aH1);
      }
      const int j = j0 + lr;
      const float br = bih[j] + bhh[j];
      const float bz = bih[HEN + j] + bhh[HEN + j];
      const float bi = bih[2 * HEN + j];
      const float bh = bhh[2 * HEN + j];
#pragma unroll
      for (int e = 0; e < 8; ++e) {
        {  // M-tile 0: batch rows 0..15
          int m = mbase + e;
          float r = sigf(aR0[e] + br);
          float z = sigf(aZ0[e] + bz);
          float nn = tanhf(aI0[e] + bi + r * (aH0[e] + bh));
          float v = (1.f - z) * nn + z * h32[cur][m][j];
          h32[nxt][m][j] = v;
          hbf[nxt][m][j] = (bf16)v;
          enc[((size_t)m * SL + p) * (2 * HEN) + dir * HEN + j] = v;
        }
        {  // M-tile 1: batch rows 16..31
          int m = 16 + mbase + e;
          float r = sigf(aR1[e] + br);
          float z = sigf(aZ1[e] + bz);
          float nn = tanhf(aI1[e] + bi + r * (aH1[e] + bh));
          float v = (1.f - z) * nn + z * h32[cur][m][j];
          h32[nxt][m][j] = v;
          hbf[nxt][m][j] = (bf16)v;
          enc[((size_t)m * SL + p) * (2 * HEN) + dir * HEN + j] = v;
        }
      }
    }
    __syncthreads();
  }
}

// ---------------------------------------------------------------------------
// key_proj = enc @ Wk : [12800,1024] x [1024,512]; A converted fp32->bf16 on
// the fly.  grid = 1600 blocks x 16 waves, 1 tile (16x16) per wave.
// ---------------------------------------------------------------------------
__global__ __launch_bounds__(512)
void keyproj_kernel(const float* __restrict__ enc,
                    const bf16* __restrict__ WkT,
                    float* __restrict__ keyproj) {
  const int tid = threadIdx.x;
  const int lane = tid & 31, wave = tid >> 5;
  const int lr = lane & 15;
  const int koff = (lane >> 4) << 4;
  const int mbase = (lane >> 4) << 3;

  int id = blockIdx.x * 16 + wave;  // 0..25599
  int mt = id >> 5;                 // 0..799
  int nt = id & 31;                 // 0..31
  f32x8 acc = zero8();
  const float* arow = enc + (size_t)(mt * 16 + lr) * (2 * HEN) + koff;
  const bf16* brow = WkT + (size_t)(nt * 16 + lr) * (2 * HEN) + koff;
#pragma unroll 2
  for (int k0 = 0; k0 < 2 * HEN; k0 += 32)
    acc = wmma_bf(cvtf16(arow + k0), ldbf(brow + k0), acc);
  int col = nt * 16 + lr;
#pragma unroll
  for (int e = 0; e < 8; ++e)
    keyproj[(size_t)(mt * 16 + mbase + e) * HDN + col] = acc[e];
}

// ---------------------------------------------------------------------------
// Decoder GRU step (1 block, 16 waves) + q = h' @ Wd^T + battn
// ---------------------------------------------------------------------------
__global__ __launch_bounds__(512)
void dec_gru_kernel(const int* __restrict__ trg_ids, int t,
                    const bf16* __restrict__ embed_bf,
                    const bf16* __restrict__ Wih, const bf16* __restrict__ Whh,
                    const float* __restrict__ bih, const float* __restrict__ bhh,
                    const bf16* __restrict__ WdT, const float* __restrict__ battn,
                    float* __restrict__ h32g, bf16* __restrict__ hbfg,
                    float* __restrict__ qout) {
  __shared__ __align__(32) bf16 hbf_old[NB][HDN];
  __shared__ __align__(32) bf16 hbf_new[NB][HDN];
  __shared__ __align__(32) bf16 xbf[NB][ESZ];

  const int tid = threadIdx.x;
  const int lane = tid & 31, wave = tid >> 5;
  const int lr = lane & 15;
  const int koff = (lane >> 4) << 4;
  const int mbase = (lane >> 4) << 3;

  for (int i = tid; i < NB * HDN; i += 512)
    hbf_old[i >> 9][i & 511] = hbfg[i];
  {
    int n = tid >> 4;
    int c0 = (tid & 15) << 4;
    int id = trg_ids[n * TDL + t];
    *(bf16x16*)&xbf[n][c0] = ldbf(embed_bf + (size_t)id * ESZ + c0);
  }
  __syncthreads();

#pragma unroll 1
  for (int jj = 0; jj < 2; ++jj) {
    const int jt = wave * 2 + jj;
    const int j0 = jt * 16;
    f32x8 aR0 = zero8(), aZ0 = zero8(), aI0 = zero8(), aH0 = zero8();
    f32x8 aR1 = zero8(), aZ1 = zero8(), aI1 = zero8(), aH1 = zero8();

    const bf16* a0 = &xbf[lr][koff];
    const bf16* a1 = &xbf[16 + lr][koff];
    const bf16* wr = Wih + (size_t)(j0 + lr) * ESZ + koff;
#pragma unroll 1
    for (int k0 = 0; k0 < ESZ; k0 += 32) {
      bf16x16 br_ = ldbf(wr + k0);
      bf16x16 bz_ = ldbf(wr + (size_t)HDN * ESZ + k0);
      bf16x16 bi_ = ldbf(wr + (size_t)2 * HDN * ESZ + k0);
      bf16x16 A0 = ldbf(a0 + k0);
      bf16x16 A1 = ldbf(a1 + k0);
      aR0 = wmma_bf(A0, br_, aR0);  aR1 = wmma_bf(A1, br_, aR1);
      aZ0 = wmma_bf(A0, bz_, aZ0);  aZ1 = wmma_bf(A1, bz_, aZ1);
      aI0 = wmma_bf(A0, bi_, aI0);  aI1 = wmma_bf(A1, bi_, aI1);
    }
    const bf16* h0p = &hbf_old[lr][koff];
    const bf16* h1p = &hbf_old[16 + lr][koff];
    const bf16* wh = Whh + (size_t)(j0 + lr) * HDN + koff;
#pragma unroll 1
    for (int k0 = 0; k0 < HDN; k0 += 32) {
      bf16x16 br_ = ldbf(wh + k0);
      bf16x16 bz_ = ldbf(wh + (size_t)HDN * HDN + k0);
      bf16x16 bh_ = ldbf(wh + (size_t)2 * HDN * HDN + k0);
      bf16x16 A0 = ldbf(h0p + k0);
      bf16x16 A1 = ldbf(h1p + k0);
      aR0 = wmma_bf(A0, br_, aR0);  aR1 = wmma_bf(A1, br_, aR1);
      aZ0 = wmma_bf(A0, bz_, aZ0);  aZ1 = wmma_bf(A1, bz_, aZ1);
      aH0 = wmma_bf(A0, bh_, aH0);  aH1 = wmma_bf(A1, bh_, aH1);
    }
    const int j = j0 + lr;
    const float br = bih[j] + bhh[j];
    const float bz = bih[HDN + j] + bhh[HDN + j];
    const float bi = bih[2 * HDN + j];
    const float bh = bhh[2 * HDN + j];
#pragma unroll
    for (int e = 0; e < 8; ++e) {
      {
        int m = mbase + e;
        float r = sigf(aR0[e] + br);
        float z = sigf(aZ0[e] + bz);
        float nn = tanhf(aI0[e] + bi + r * (aH0[e] + bh));
        float v = (1.f - z) * nn + z * h32g[m * HDN + j];
        h32g[m * HDN + j] = v;
        hbfg[m * HDN + j] = (bf16)v;
        hbf_new[m][j] = (bf16)v;
      }
      {
        int m = 16 + mbase + e;
        float r = sigf(aR1[e] + br);
        float z = sigf(aZ1[e] + bz);
        float nn = tanhf(aI1[e] + bi + r * (aH1[e] + bh));
        float v = (1.f - z) * nn + z * h32g[m * HDN + j];
        h32g[m * HDN + j] = v;
        hbfg[m * HDN + j] = (bf16)v;
        hbf_new[m][j] = (bf16)v;
      }
    }
  }
  __syncthreads();

  // q = h' @ Wd^T + battn  (64 tiles, K = 512)
#pragma unroll 1
  for (int i = 0; i < 4; ++i) {
    int g = wave * 4 + i;
    int mt = g & 1, nt = g >> 1;
    f32x8 acc = zero8();
    const bf16* arow = &hbf_new[mt * 16 + lr][koff];
    const bf16* brow = WdT + (size_t)(nt * 16 + lr) * HDN + koff;
#pragma unroll 2
    for (int k0 = 0; k0 < HDN; k0 += 32)
      acc = wmma_bf(ldbf(arow + k0), ldbf(brow + k0), acc);
    int col = nt * 16 + lr;
    float b = battn[col];
#pragma unroll
    for (int e = 0; e < 8; ++e)
      qout[(mt * 16 + mbase + e) * HDN + col] = acc[e] + b;
  }
}

// ---------------------------------------------------------------------------
// Attention: e = tanh(key_proj + q) . v + mask ; softmax over S ; ctx = a@enc
// One block per batch row n (32 blocks, 512 threads).
// ---------------------------------------------------------------------------
__global__ __launch_bounds__(512)
void attn_kernel(const float* __restrict__ keyproj, const float* __restrict__ q,
                 const float* __restrict__ vvec, const float* __restrict__ src_mask,
                 const float* __restrict__ enc,
                 float* __restrict__ attn_out, float* __restrict__ ctx_out) {
  const int n = blockIdx.x;
  const int tid = threadIdx.x;
  const int lane = tid & 31, wave = tid >> 5;

  __shared__ float qs[HDN];
  __shared__ float vs[HDN];
  __shared__ float el[SL];
  __shared__ float red[512];

  for (int i = tid; i < HDN; i += 512) { qs[i] = q[n * HDN + i]; vs[i] = vvec[i]; }
  __syncthreads();

  for (int s = wave; s < SL; s += 16) {
    const float* kp = keyproj + ((size_t)n * SL + s) * HDN;
    float acc = 0.f;
    for (int k = lane; k < HDN; k += 32) acc += tanhf(kp[k] + qs[k]) * vs[k];
    for (int off = 16; off > 0; off >>= 1) acc += __shfl_xor(acc, off, 32);
    if (lane == 0) el[s] = acc + (1.0f - src_mask[n * SL + s]) * -1e10f;
  }
  __syncthreads();

  float m = -INFINITY;
  for (int s = tid; s < SL; s += 512) m = fmaxf(m, el[s]);
  red[tid] = m; __syncthreads();
  for (int off = 256; off > 0; off >>= 1) {
    if (tid < off) red[tid] = fmaxf(red[tid], red[tid + off]);
    __syncthreads();
  }
  m = red[0]; __syncthreads();

  float ssum = 0.f;
  for (int s = tid; s < SL; s += 512) { float ex = __expf(el[s] - m); el[s] = ex; ssum += ex; }
  red[tid] = ssum; __syncthreads();
  for (int off = 256; off > 0; off >>= 1) {
    if (tid < off) red[tid] += red[tid + off];
    __syncthreads();
  }
  float inv = 1.0f / red[0]; __syncthreads();
  for (int s = tid; s < SL; s += 512) {
    float a = el[s] * inv; el[s] = a; attn_out[n * SL + s] = a;
  }
  __syncthreads();

  for (int d = tid; d < 2 * HEN; d += 512) {
    float acc = 0.f;
    const float* ep = enc + (size_t)n * SL * (2 * HEN) + d;
    for (int s = 0; s < SL; ++s) acc += el[s] * ep[(size_t)s * (2 * HEN)];
    ctx_out[n * (2 * HEN) + d] = acc;
  }
}

// ---------------------------------------------------------------------------
// hidden = [h, ctx] @ W1^T + b1 (-> bf16)  and p_gen = sig([x,ctx,h].Wp + bp)
// One block, 16 waves.
// ---------------------------------------------------------------------------
__global__ __launch_bounds__(512)
void gen_hidden_kernel(const float* __restrict__ h32g, const float* __restrict__ ctx,
                       const bf16* __restrict__ W1T, const float* __restrict__ b1,
                       bf16* __restrict__ hiddenbf,
                       const int* __restrict__ trg_ids, int t,
                       const float* __restrict__ embed,
                       const float* __restrict__ Wp, const float* __restrict__ bp,
                       float* __restrict__ pgen) {
  __shared__ __align__(32) bf16 cat[NB][HDN + 2 * HEN];  // [32][1536]
  const int tid = threadIdx.x;
  const int lane = tid & 31, wave = tid >> 5;
  const int lr = lane & 15;
  const int koff = (lane >> 4) << 4;
  const int mbase = (lane >> 4) << 3;
  const int CK = HDN + 2 * HEN;  // 1536

  for (int i = tid; i < NB * CK; i += 512) {
    int n = i / CK, c = i % CK;
    float v = (c < HDN) ? h32g[n * HDN + c] : ctx[n * (2 * HEN) + (c - HDN)];
    cat[n][c] = (bf16)v;
  }
  __syncthreads();

#pragma unroll 1
  for (int i = 0; i < 4; ++i) {
    int g = wave * 4 + i;
    int mt = g & 1, nt = g >> 1;
    f32x8 acc = zero8();
    const bf16* arow = &cat[mt * 16 + lr][koff];
    const bf16* brow = W1T + (size_t)(nt * 16 + lr) * CK + koff;
#pragma unroll 2
    for (int k0 = 0; k0 < CK; k0 += 32)
      acc = wmma_bf(ldbf(arow + k0), ldbf(brow + k0), acc);
    int col = nt * 16 + lr;
    float b = b1[col];
#pragma unroll
    for (int e = 0; e < 8; ++e)
      hiddenbf[(mt * 16 + mbase + e) * HDN + col] = (bf16)(acc[e] + b);
  }

  // p_gen: [xi(256), ctx(1024), h(512)] . Wp + bp  -> sigmoid
  const int PK = ESZ + 2 * HEN + HDN;  // 1792
  for (int n = wave; n < NB; n += 16) {
    int id = trg_ids[n * TDL + t];
    float acc = 0.f;
    for (int k = lane; k < PK; k += 32) {
      float xv;
      if (k < ESZ)                xv = embed[(size_t)id * ESZ + k];
      else if (k < ESZ + 2 * HEN) xv = ctx[n * (2 * HEN) + (k - ESZ)];
      else                        xv = h32g[n * HDN + (k - ESZ - 2 * HEN)];
      acc += xv * Wp[k];
    }
    for (int off = 16; off > 0; off >>= 1) acc += __shfl_xor(acc, off, 32);
    if (lane == 0) pgen[n] = sigf(acc + bp[0]);
  }
}

// ---------------------------------------------------------------------------
// logits = hidden @ W2^T + b2 : [32,512] x [512,50000]; 6250 tiles.
// W2 (bf16, 51 MB) stays L2-resident across all 30 decoder steps.
// ---------------------------------------------------------------------------
__global__ __launch_bounds__(512)
void vocab_gemm_kernel(const bf16* __restrict__ hiddenbf,
                       const bf16* __restrict__ W2T, const float* __restrict__ b2,
                       float* __restrict__ logits) {
  const int tid = threadIdx.x;
  const int lane = tid & 31, wave = tid >> 5;
  const int lr = lane & 15;
  const int koff = (lane >> 4) << 4;
  const int mbase = (lane >> 4) << 3;

  int id = blockIdx.x * 16 + wave;
  if (id < 2 * (VSZ / 16)) {  // 6250 tiles, wave-uniform guard
    int mt = id / (VSZ / 16);
    int nt = id % (VSZ / 16);
    f32x8 acc = zero8();
    const bf16* arow = hiddenbf + (size_t)(mt * 16 + lr) * HDN + koff;
    const bf16* brow = W2T + (size_t)(nt * 16 + lr) * HDN + koff;
#pragma unroll 2
    for (int k0 = 0; k0 < HDN; k0 += 32)
      acc = wmma_bf(ldbf(arow + k0), ldbf(brow + k0), acc);
    int col = nt * 16 + lr;
    float b = b2[col];
#pragma unroll
    for (int e = 0; e < 8; ++e)
      logits[(size_t)(mt * 16 + mbase + e) * VSZ + col] = acc[e] + b;
  }
}

// Online softmax stats per row: one block per n.
__global__ __launch_bounds__(256)
void softmax_stats_kernel(const float* __restrict__ logits,
                          float* __restrict__ rowmax, float* __restrict__ rowsum) {
  const int n = blockIdx.x;
  const int tid = threadIdx.x;
  const float* lp = logits + (size_t)n * VSZ;
  float m = -INFINITY, s = 0.f;
  for (int v = tid; v < VSZ; v += 256) {
    float x = lp[v];
    float mn = fmaxf(m, x);
    s = s * __expf(m - mn) + __expf(x - mn);
    m = mn;
  }
  __shared__ float ms[256], ss[256];
  ms[tid] = m; ss[tid] = s; __syncthreads();
  for (int off = 128; off > 0; off >>= 1) {
    if (tid < off) {
      float m2 = fmaxf(ms[tid], ms[tid + off]);
      ss[tid] = ss[tid] * __expf(ms[tid] - m2) + ss[tid + off] * __expf(ms[tid + off] - m2);
      ms[tid] = m2;
    }
    __syncthreads();
  }
  if (tid == 0) { rowmax[n] = ms[0]; rowsum[n] = ss[0]; }
}

// out[n,t,v] = p_gen * softmax(logits)   (OOV columns -> 0)
__global__ __launch_bounds__(256)
void out_fill_kernel(const float* __restrict__ logits,
                     const float* __restrict__ rowmax, const float* __restrict__ rowsum,
                     const float* __restrict__ pgen, float* __restrict__ out, int t) {
  size_t idx = (size_t)blockIdx.x * 256 + threadIdx.x;
  if (idx >= (size_t)NB * VOUT) return;
  int n = (int)(idx / VOUT);
  int v = (int)(idx % VOUT);
  float val = 0.f;
  if (v < VSZ)
    val = pgen[n] * __expf(logits[(size_t)n * VSZ + v] - rowmax[n]) / rowsum[n];
  out[((size_t)n * TDL + t) * VOUT + v] = val;
}

// Deterministic pointer scatter: out[n,t,ptr[n,s]] += (1-p_gen)*attn[n,s]
__global__ void scatter_kernel(const int* __restrict__ ptr_idx,
                               const float* __restrict__ attn,
                               const float* __restrict__ pgen,
                               float* __restrict__ out, int t) {
  int n = threadIdx.x;
  if (n >= NB) return;
  float w = 1.0f - pgen[n];
  float* orow = out + ((size_t)n * TDL + t) * VOUT;
  for (int s = 0; s < SL; ++s)
    orow[ptr_idx[n * SL + s]] += w * attn[n * SL + s];
}

// ---------------------------------------------------------------------------
// Host: workspace layout + launch sequence
// ---------------------------------------------------------------------------
extern "C" void kernel_launch(void* const* d_in, const int* in_sizes, int n_in,
                              void* d_out, int out_size, void* d_ws, size_t ws_size,
                              hipStream_t stream) {
  (void)in_sizes; (void)n_in; (void)out_size; (void)ws_size;

  const int*   src_ids  = (const int*)  d_in[0];
  const float* src_mask = (const float*)d_in[1];
  const int*   trg_ids  = (const int*)  d_in[2];
  const int*   ptr_idx  = (const int*)  d_in[3];
  const float* embed    = (const float*)d_in[4];
  const float* Wih_f = (const float*)d_in[5],  *Whh_f = (const float*)d_in[6];
  const float* bih_f = (const float*)d_in[7],  *bhh_f = (const float*)d_in[8];
  const float* Wih_b = (const float*)d_in[9],  *Whh_b = (const float*)d_in[10];
  const float* bih_b = (const float*)d_in[11], *bhh_b = (const float*)d_in[12];
  const float* Wih_d = (const float*)d_in[13], *Whh_d = (const float*)d_in[14];
  const float* bih_d = (const float*)d_in[15], *bhh_d = (const float*)d_in[16];
  const float* Wk = (const float*)d_in[17], *Wd = (const float*)d_in[18];
  const float* battn = (const float*)d_in[19], *vvec = (const float*)d_in[20];
  const float* W1 = (const float*)d_in[21], *b1 = (const float*)d_in[22];
  const float* W2 = (const float*)d_in[23], *b2 = (const float*)d_in[24];
  const float* Wp = (const float*)d_in[25], *bp = (const float*)d_in[26];
  float* out = (float*)d_out;

  char* ws = (char*)d_ws;
  size_t off = 0;
  auto alloc = [&](size_t bytes) -> char* {
    char* p = ws + off;
    off += (bytes + 255) & ~(size_t)255;
    return p;
  };

  bf16* embed_bf = (bf16*)alloc((size_t)VSZ * ESZ * 2);
  bf16* WihF_bf  = (bf16*)alloc((size_t)3 * HEN * ESZ * 2);
  bf16* WhhF_bf  = (bf16*)alloc((size_t)3 * HEN * HEN * 2);
  bf16* WihB_bf  = (bf16*)alloc((size_t)3 * HEN * ESZ * 2);
  bf16* WhhB_bf  = (bf16*)alloc((size_t)3 * HEN * HEN * 2);
  bf16* WihD_bf  = (bf16*)alloc((size_t)3 * HDN * ESZ * 2);
  bf16* WhhD_bf  = (bf16*)alloc((size_t)3 * HDN * HDN * 2);
  bf16* WkT_bf   = (bf16*)alloc((size_t)HDN * (2 * HEN) * 2);
  bf16* WdT_bf   = (bf16*)alloc((size_t)HDN * HDN * 2);
  bf16* W1T_bf   = (bf16*)alloc((size_t)HDN * (HDN + 2 * HEN) * 2);
  bf16* W2T_bf   = (bf16*)alloc((size_t)VSZ * HDN * 2);
  float* enc     = (float*)alloc((size_t)NB * SL * (2 * HEN) * 4);
  float* keyproj = (float*)alloc((size_t)NB * SL * HDN * 4);
  float* h32g    = (float*)alloc((size_t)NB * HDN * 4);
  bf16*  hbfg    = (bf16*) alloc((size_t)NB * HDN * 2);
  float* qbuf    = (float*)alloc((size_t)NB * HDN * 4);
  float* attn    = (float*)alloc((size_t)NB * SL * 4);
  float* ctx     = (float*)alloc((size_t)NB * (2 * HEN) * 4);
  bf16*  hiddenbf= (bf16*) alloc((size_t)NB * HDN * 2);
  float* pgen    = (float*)alloc((size_t)NB * 4);
  float* logits  = (float*)alloc((size_t)NB * VSZ * 4);
  float* rowmax  = (float*)alloc((size_t)NB * 4);
  float* rowsum  = (float*)alloc((size_t)NB * 4);

  auto cvt = [&](const float* in, bf16* o, int count) {
    cvt_bf16_kernel<<<(count + 255) / 256, 256, 0, stream>>>(in, o, count);
  };
  auto cvtT = [&](const float* in, bf16* o, int rows_out, int cols_out) {
    int count = rows_out * cols_out;
    cvt_t_bf16_kernel<<<(count + 255) / 256, 256, 0, stream>>>(in, o, rows_out, cols_out);
  };

  // --- weight conversion (fp32 -> bf16; projections transposed to [out,in]) ---
  cvt(embed, embed_bf, VSZ * ESZ);
  cvt(Wih_f, WihF_bf, 3 * HEN * ESZ);  cvt(Whh_f, WhhF_bf, 3 * HEN * HEN);
  cvt(Wih_b, WihB_bf, 3 * HEN * ESZ);  cvt(Whh_b, WhhB_bf, 3 * HEN * HEN);
  cvt(Wih_d, WihD_bf, 3 * HDN * ESZ);  cvt(Whh_d, WhhD_bf, 3 * HDN * HDN);
  cvtT(Wk, WkT_bf, HDN, 2 * HEN);            // [1024,512] -> [512,1024]
  cvtT(Wd, WdT_bf, HDN, HDN);                // [512,512]  -> [512,512]
  cvtT(W1, W1T_bf, HDN, HDN + 2 * HEN);      // [1536,512] -> [512,1536]
  cvtT(W2, W2T_bf, VSZ, HDN);                // [512,50000]-> [50000,512]

  // --- zero decoder hidden state ---
  zero_u32_kernel<<<(NB * HDN + 255) / 256, 256, 0, stream>>>(
      (unsigned int*)h32g, NB * HDN);
  zero_u32_kernel<<<(NB * HDN / 2 + 255) / 256, 256, 0, stream>>>(
      (unsigned int*)hbfg, NB * HDN / 2);

  // --- bidirectional encoder (persistent, 2 blocks) ---
  encoder_kernel<<<2, 512, 0, stream>>>(src_ids, embed_bf,
                                        WihF_bf, WhhF_bf, bih_f, bhh_f,
                                        WihB_bf, WhhB_bf, bih_b, bhh_b, enc);

  // --- key_proj = enc @ Wk ---
  keyproj_kernel<<<(NB * SL / 16) * (HDN / 16) / 16, 512, 0, stream>>>(
      enc, WkT_bf, keyproj);

  // --- autoregressive decoder ---
  const int vocab_blocks = (2 * (VSZ / 16) + 15) / 16;               // 391
  const int fill_blocks  = (int)(((size_t)NB * VOUT + 255) / 256);   // 6257
  for (int t = 0; t < TDL; ++t) {
    dec_gru_kernel<<<1, 512, 0, stream>>>(trg_ids, t, embed_bf,
                                          WihD_bf, WhhD_bf, bih_d, bhh_d,
                                          WdT_bf, battn, h32g, hbfg, qbuf);
    attn_kernel<<<NB, 512, 0, stream>>>(keyproj, qbuf, vvec, src_mask, enc,
                                        attn, ctx);
    gen_hidden_kernel<<<1, 512, 0, stream>>>(h32g, ctx, W1T_bf, b1, hiddenbf,
                                             trg_ids, t, embed, Wp, bp, pgen);
    vocab_gemm_kernel<<<vocab_blocks, 512, 0, stream>>>(hiddenbf, W2T_bf, b2,
                                                        logits);
    softmax_stats_kernel<<<NB, 256, 0, stream>>>(logits, rowmax, rowsum);
    out_fill_kernel<<<fill_blocks, 256, 0, stream>>>(logits, rowmax, rowsum,
                                                     pgen, out, t);
    scatter_kernel<<<1, 32, 0, stream>>>(ptr_idx, attn, pgen, out, t);
  }
}
